// MultiHeadedAttention_74586402062438
// MI455X (gfx1250) — compile-verified, matrix operand order checked by reading, not run on hardware
//
#include <hip/hip_runtime.h>
#include <hip/hip_bf16.h>

typedef __attribute__((ext_vector_type(16))) _Float16 v16h;
typedef __attribute__((ext_vector_type(8)))  _Float16 v8h;
typedef __attribute__((ext_vector_type(8)))  float    v8f;
typedef __attribute__((ext_vector_type(4)))  float    v4f;
typedef __attribute__((ext_vector_type(4)))  int      v4i;

// Problem constants (match reference): B=8, S=1024, D=1024, H=16, DK=64
#define BB   8
#define SS   1024
#define DD   1024
#define HH   16
#define DKK  64

// ---------------------------------------------------------------------------
// gfx1250 async global->LDS copy (ASYNCcnt-tracked), guarded so compilation
// can never regress if the builtin is absent on this toolchain.
// Builtin signature (from hipcc diagnostic): first param is AS1 (printed as
// "__device__") pointer to int4; LDS side is AS3 pointer to int4.
// ---------------------------------------------------------------------------
#if __has_builtin(__builtin_amdgcn_global_load_async_to_lds_b128)
#define HAVE_ASYNC_LDS 1

__device__ __forceinline__ void async_copy16(const void* gsrc, void* ldst) {
  __attribute__((address_space(1))) v4i* g =
      (__attribute__((address_space(1))) v4i*)(unsigned long long)gsrc;
  // generic->AS3: low 32 bits of a generic LDS pointer are the LDS offset
  __attribute__((address_space(3))) v4i* l =
      (__attribute__((address_space(3))) v4i*)(unsigned int)(unsigned long long)ldst;
  __builtin_amdgcn_global_load_async_to_lds_b128(g, l, 0, 0);
}
__device__ __forceinline__ void async_wait0() {
#if __has_builtin(__builtin_amdgcn_s_wait_asynccnt)
  __builtin_amdgcn_s_wait_asynccnt(0);
#else
  asm volatile("s_wait_asynccnt 0x0" ::: "memory");
#endif
}
#endif

// ---------------------------------------------------------------------------
// WMMA helpers
// ---------------------------------------------------------------------------
__device__ __forceinline__ v8f wmma_f16(v16h a, v16h b, v8f c) {
  // D = A(16x32 f16) * B(32x16 f16) + C(16x16 f32)
  return __builtin_amdgcn_wmma_f32_16x16x32_f16(false, a, false, b,
                                                (short)0, c, false, false);
}

// Load one 16x32 f16 fragment (A-layout / B-column layout) from row-major
// storage. base -> element (row 0, k 0); strideHalves = row pitch in halves
// (must be a multiple of 8 for 16B-aligned b128 accesses).
// ISA layout: lane<16 holds row (lane&15), K = 0..7 and 16..23;
//             lane>=16 holds same row,    K = 8..15 and 24..31.
__device__ __forceinline__ v16h load_frag(const _Float16* __restrict__ base,
                                          int strideHalves, int lane) {
  const _Float16* p = base + (size_t)(lane & 15) * strideHalves + ((lane & 16) ? 8 : 0);
  v8h lo = *(const v8h*)(p);
  v8h hi = *(const v8h*)(p + 16);
  return __builtin_shufflevector(lo, hi, 0, 1, 2, 3, 4, 5, 6, 7,
                                 8, 9, 10, 11, 12, 13, 14, 15);
}

// ---------------------------------------------------------------------------
// GEMM: C[M=8192, N=1024] = A @ W^T + bias
//   mode 0: A = f32 activations, dst = f16 head-major [B,H,S,DK] (QKV proj)
//   mode 1: A = f16 heads buffer, dst = f32 flat [M,1024]        (out proj)
// Block: 256 thr (8 waves), tile 128x128, K-step 32.
// Wave (wm=w>>1, wn=w&1) computes a 32x64 sub-tile = 2x4 WMMA accumulators.
// ---------------------------------------------------------------------------
__global__ __launch_bounds__(256)
void mha_gemm_kernel(const void* __restrict__ Aab,
                     const float* __restrict__ W,
                     const float* __restrict__ bias,
                     void* __restrict__ dst, int mode) {
  __shared__ alignas(16) _Float16 As[128][40];
  __shared__ alignas(16) _Float16 Bs[128][40];

  const int tid  = threadIdx.x;
  const int lane = tid & 31;
  const int w    = tid >> 5;
  const int wm   = w >> 1;   // 0..3
  const int wn   = w & 1;    // 0..1
  const int m0   = blockIdx.x * 128;
  const int n0   = blockIdx.y * 128;

  v8f acc[2][4];
#pragma unroll
  for (int i = 0; i < 2; ++i)
#pragma unroll
    for (int j = 0; j < 4; ++j) { v8f z = {}; acc[i][j] = z; }

  const int ldrow = tid >> 1;          // 0..127
  const int ldseg = (tid & 1) * 16;    // 0 or 16

  for (int kk = 0; kk < DD; kk += 32) {
    // ---- stage A tile (128x32) ----
    if (mode == 0) {
      const float* ap = (const float*)Aab + (size_t)(m0 + ldrow) * DD + kk + ldseg;
      const v4f* p4 = (const v4f*)ap;
      v4f f0 = p4[0], f1 = p4[1], f2 = p4[2], f3 = p4[3];
      v8h h0, h1;
#pragma unroll
      for (int i = 0; i < 4; ++i) {
        h0[i]     = (_Float16)f0[i];  h0[4 + i] = (_Float16)f1[i];
        h1[i]     = (_Float16)f2[i];  h1[4 + i] = (_Float16)f3[i];
      }
      *(v8h*)&As[ldrow][ldseg]     = h0;
      *(v8h*)&As[ldrow][ldseg + 8] = h1;
      if (kk + 32 < DD) __builtin_prefetch(ap + 32, 0, 3);
    } else {
      const _Float16* ap = (const _Float16*)Aab + (size_t)(m0 + ldrow) * DD + kk + ldseg;
#if defined(HAVE_ASYNC_LDS)
      async_copy16(ap,     &As[ldrow][ldseg]);
      async_copy16(ap + 8, &As[ldrow][ldseg + 8]);
#else
      *(v8h*)&As[ldrow][ldseg]     = *(const v8h*)(ap);
      *(v8h*)&As[ldrow][ldseg + 8] = *(const v8h*)(ap + 8);
#endif
      if (kk + 32 < DD) __builtin_prefetch(ap + 32, 0, 3);
    }
    // ---- stage W tile (128 rows of W = 128 columns of B) ----
    {
      const float* wp = W + (size_t)(n0 + ldrow) * DD + kk + ldseg;
      const v4f* p4 = (const v4f*)wp;
      v4f f0 = p4[0], f1 = p4[1], f2 = p4[2], f3 = p4[3];
      v8h h0, h1;
#pragma unroll
      for (int i = 0; i < 4; ++i) {
        h0[i]     = (_Float16)f0[i];  h0[4 + i] = (_Float16)f1[i];
        h1[i]     = (_Float16)f2[i];  h1[4 + i] = (_Float16)f3[i];
      }
      *(v8h*)&Bs[ldrow][ldseg]     = h0;
      *(v8h*)&Bs[ldrow][ldseg + 8] = h1;
      if (kk + 32 < DD) __builtin_prefetch(wp + 32, 0, 3);
    }
#if defined(HAVE_ASYNC_LDS)
    if (mode != 0) async_wait0();   // drain ASYNCcnt before workgroup barrier
#endif
    __syncthreads();

    v16h af[2], bf[4];
#pragma unroll
    for (int i = 0; i < 2; ++i)
      af[i] = load_frag(&As[wm * 32 + i * 16][0], 40, lane);
#pragma unroll
    for (int j = 0; j < 4; ++j)
      bf[j] = load_frag(&Bs[wn * 64 + j * 16][0], 40, lane);
#pragma unroll
    for (int i = 0; i < 2; ++i)
#pragma unroll
      for (int j = 0; j < 4; ++j)
        acc[i][j] = wmma_f16(af[i], bf[j], acc[i][j]);
    __syncthreads();
  }

  // ---- epilogue: C layout -> lane<16: (row=j, col=lane); lane>=16: (row=j+8)
  const int rhalf = (lane & 16) ? 8 : 0;
  const int cn    = lane & 15;
#pragma unroll
  for (int i = 0; i < 2; ++i) {
#pragma unroll
    for (int j = 0; j < 4; ++j) {
      const int ncol = n0 + wn * 64 + j * 16 + cn;
      const float bv = bias[ncol];
#pragma unroll
      for (int r = 0; r < 8; ++r) {
        const int grow = m0 + wm * 32 + i * 16 + r + rhalf;   // 0..8191 = b*S+s
        const float val = acc[i][j][r] + bv;
        if (mode == 0) {
          const int b = grow >> 10, s = grow & 1023;
          const int h = ncol >> 6,  d = ncol & 63;
          ((_Float16*)dst)[(((size_t)(b * HH + h) * SS + s) << 6) + d] = (_Float16)val;
        } else {
          ((float*)dst)[(size_t)grow * DD + ncol] = val;
        }
      }
    }
  }
}

// ---------------------------------------------------------------------------
// Flash attention with zero-filled (NOT -inf) masking, per reference:
//   a = where(causal, scores/8, 0); softmax over the FULL row (masked entries
//   still contribute exp(0-max) mass) -> run online softmax over all K tiles
//   with masked scores replaced by 0.0.
// Grid: (S/128, B*H). Block 256 thr = 8 waves; wave w handles 16 query rows.
// ---------------------------------------------------------------------------
__global__ __launch_bounds__(256)
void mha_attn_kernel(const _Float16* __restrict__ Q,
                     const _Float16* __restrict__ K,
                     const _Float16* __restrict__ V,
                     _Float16* __restrict__ Hd) {
  __shared__ alignas(16) _Float16 Ks[32][64];      // K tile, row-major
  __shared__ alignas(16) _Float16 Vt[64][40];      // V tile transposed [dk][k]
  __shared__ alignas(16) _Float16 Ps[8][16][32];   // per-wave P scratch

  const int tid   = threadIdx.x;
  const int lane  = tid & 31;
  const int w     = tid >> 5;
  const int bh    = blockIdx.y;             // b*H + h
  const int qbase = blockIdx.x * 128;
  const int q0    = qbase + w * 16;
  const int rhalf = (lane & 16) ? 8 : 0;
  const int cn    = lane & 15;

  const _Float16* Qp = Q + ((size_t)bh * SS + q0) * DKK;
  const _Float16* Kp = K + (size_t)bh * SS * DKK;
  const _Float16* Vp = V + (size_t)bh * SS * DKK;

  // Q fragments (16 rows x 64 dk = two K-steps), held in registers
  v16h aq0 = load_frag(Qp, DKK, lane);
  v16h aq1 = load_frag(Qp + 32, DKK, lane);

  float rmax[8], rsum[8];
  v8f ov[4];
#pragma unroll
  for (int j = 0; j < 8; ++j) { rmax[j] = -1e30f; rsum[j] = 0.0f; }
#pragma unroll
  for (int t = 0; t < 4; ++t) { v8f z = {}; ov[t] = z; }

  const float scale = 0.125f;  // 1/sqrt(64)

  for (int kk = 0; kk < SS; kk += 32) {
    // ---- stage K (async row-major copy) and V (transposed) tiles ----
    {
      const int r = tid >> 3;            // 0..31
      const int cseg = (tid & 7) * 8;    // 0..56
      const _Float16* kg = Kp + (size_t)(kk + r) * DKK + cseg;
      const _Float16* vg = Vp + (size_t)(kk + r) * DKK + cseg;
#if defined(HAVE_ASYNC_LDS)
      async_copy16(kg, &Ks[r][cseg]);    // ASYNCcnt-tracked, no VGPR staging
#else
      *(v8h*)&Ks[r][cseg] = *(const v8h*)kg;
#endif
      v8h vv = *(const v8h*)vg;
#pragma unroll
      for (int i = 0; i < 8; ++i) Vt[cseg + i][r] = vv[i];
      if (kk + 32 < SS) {
        __builtin_prefetch(kg + 32 * DKK, 0, 3);
        __builtin_prefetch(vg + 32 * DKK, 0, 3);
      }
#if defined(HAVE_ASYNC_LDS)
      async_wait0();                     // drain before the barrier
#endif
    }
    __syncthreads();

    // ---- scores: S[16q x 32k] via 4 WMMAs (two 16-col n-tiles, dk=64) ----
    v8f s0 = {}, s1 = {};
    s0 = wmma_f16(aq0, load_frag(&Ks[0][0],   DKK, lane), s0);
    s0 = wmma_f16(aq1, load_frag(&Ks[0][32],  DKK, lane), s0);
    s1 = wmma_f16(aq0, load_frag(&Ks[16][0],  DKK, lane), s1);
    s1 = wmma_f16(aq1, load_frag(&Ks[16][32], DKK, lane), s1);

    // ---- online softmax (zero-filled mask), P -> LDS in A-layout order ----
#pragma unroll
    for (int j = 0; j < 8; ++j) {
      const int qg = q0 + j + rhalf;
      const int k0 = kk + cn, k1 = kk + 16 + cn;
      float x0 = (k0 <= qg) ? s0[j] * scale : 0.0f;
      float x1 = (k1 <= qg) ? s1[j] * scale : 0.0f;
      float tmax = fmaxf(x0, x1);
#pragma unroll
      for (int off = 1; off < 16; off <<= 1)
        tmax = fmaxf(tmax, __shfl_xor(tmax, off, 32));
      const float mnew = fmaxf(rmax[j], tmax);
      const float corr = __expf(rmax[j] - mnew);
      const float p0 = __expf(x0 - mnew);
      const float p1 = __expf(x1 - mnew);
      float ps = p0 + p1;
#pragma unroll
      for (int off = 1; off < 16; off <<= 1)
        ps += __shfl_xor(ps, off, 32);
      rsum[j] = rsum[j] * corr + ps;
      rmax[j] = mnew;
      const int r = j + rhalf;
      Ps[w][r][cn]      = (_Float16)p0;
      Ps[w][r][16 + cn] = (_Float16)p1;
#pragma unroll
      for (int t = 0; t < 4; ++t) ov[t][j] *= corr;
    }

    // ---- O += P(16x32) x V(32x64): 4 WMMAs ----
    v16h pa = load_frag(&Ps[w][0][0], 32, lane);
#pragma unroll
    for (int t = 0; t < 4; ++t)
      ov[t] = wmma_f16(pa, load_frag(&Vt[t * 16][0], 40, lane), ov[t]);
    __syncthreads();
  }

  // ---- normalize and store heads (f16, flat [B,S,H*DK] row-major) ----
  const int h = bh & (HH - 1);
  const int b = bh >> 4;
#pragma unroll
  for (int j = 0; j < 8; ++j) {
    const float inv = 1.0f / rsum[j];
    const int qg = q0 + j + rhalf;
    const size_t rowoff = ((size_t)b * SS + qg) * DD + (size_t)h * DKK;
#pragma unroll
    for (int t = 0; t < 4; ++t)
      Hd[rowoff + t * 16 + cn] = (_Float16)(ov[t][j] * inv);
  }
}

// ---------------------------------------------------------------------------
extern "C" void kernel_launch(void* const* d_in, const int* in_sizes, int n_in,
                              void* d_out, int out_size, void* d_ws, size_t ws_size,
                              hipStream_t stream) {
  (void)in_sizes; (void)n_in; (void)out_size; (void)ws_size;

  const float* q  = (const float*)d_in[0];
  const float* k  = (const float*)d_in[1];
  const float* v  = (const float*)d_in[2];
  // d_in[3] = causal mask: reproduced analytically (k <= q) in the kernel.
  const float* Wq = (const float*)d_in[4];
  const float* bq = (const float*)d_in[5];
  const float* Wk = (const float*)d_in[6];
  const float* bk = (const float*)d_in[7];
  const float* Wv = (const float*)d_in[8];
  const float* bv = (const float*)d_in[9];
  const float* Wo = (const float*)d_in[10];
  const float* bo = (const float*)d_in[11];

  char* ws = (char*)d_ws;
  const size_t SEG = (size_t)BB * SS * DD * sizeof(_Float16);  // 16 MiB
  _Float16* Q16  = (_Float16*)(ws);            // [B,H,S,DK] f16
  _Float16* K16  = (_Float16*)(ws + SEG);      // [B,H,S,DK] f16
  _Float16* V16  = (_Float16*)(ws + 2 * SEG);  // [B,H,S,DK] f16
  _Float16* Hd16 = (_Float16*)(ws + 3 * SEG);  // [B,S,D]    f16

  dim3 gemmGrid(BB * SS / 128, DD / 128);   // (64, 8)
  dim3 attnGrid(SS / 128, BB * HH);         // (8, 128)

  mha_gemm_kernel<<<gemmGrid, 256, 0, stream>>>(q, Wq, bq, Q16, 0);
  mha_gemm_kernel<<<gemmGrid, 256, 0, stream>>>(k, Wk, bk, K16, 0);
  mha_gemm_kernel<<<gemmGrid, 256, 0, stream>>>(v, Wv, bv, V16, 0);
  mha_attn_kernel<<<attnGrid, 256, 0, stream>>>(Q16, K16, V16, Hd16);
  mha_gemm_kernel<<<gemmGrid, 256, 0, stream>>>(Hd16, Wo, bo, d_out, 1);
}
